// RGCNLayer_22436909155027
// MI455X (gfx1250) — matrix operand order repaired and checked by reading, not compiled, not run on hardware
//
#include <hip/hip_runtime.h>

typedef __attribute__((ext_vector_type(16))) _Float16 v16h;
typedef __attribute__((ext_vector_type(8)))  _Float16 v8h;
typedef __attribute__((ext_vector_type(8)))  float    v8f;
typedef __attribute__((ext_vector_type(4)))  float    v4f;
typedef __attribute__((ext_vector_type(4)))  int      v4i;

#define R_  8
#define B_  4
#define N_  1024
#define I_  64
#define O_  64
#define NB_ 4
#define EPS 1e-7f

// ---------------------------------------------------------------------------
// K0: basis decomposition.  wfullT[r][o][i] = sum_nb w_cmp[r,nb]*weight[nb,i,o]
//     (stored transposed+f16 so WMMA B-operand reads are contiguous)
//     WoT[o][i] = Wo[i][o] in f16.
// ---------------------------------------------------------------------------
__global__ void k_wprep(const float* __restrict__ weight, const float* __restrict__ Wo,
                        const float* __restrict__ w_cmp,
                        _Float16* __restrict__ wfullT, _Float16* __restrict__ WoT) {
    int tid = blockIdx.x * 256 + threadIdx.x;       // 0 .. 36863
    if (tid < R_ * O_ * I_) {
        int r = tid >> 12, rem = tid & 4095, o = rem >> 6, i = rem & 63;
        float acc = 0.f;
#pragma unroll
        for (int nb = 0; nb < NB_; ++nb)
            acc += w_cmp[r * NB_ + nb] * weight[(nb * I_ + i) * O_ + o];
        wfullT[(r * O_ + o) * I_ + i] = (_Float16)acc;
    } else if (tid < R_ * O_ * I_ + O_ * I_) {
        int rem = tid - R_ * O_ * I_, o = rem >> 6, i = rem & 63;
        WoT[o * I_ + i] = (_Float16)Wo[i * O_ + o];
    }
}

// ---------------------------------------------------------------------------
// K1: deg[r,b,n] = sum_n' adj[r,b,n,n'].  One wave per row, coalesced b128
//     loads, wave32 shuffle reduction.  Streams adj into L2 for K3.
// ---------------------------------------------------------------------------
__global__ void k_deg(const int* __restrict__ adj, float* __restrict__ deg) {
    int lane = threadIdx.x & 31;
    int wave = threadIdx.x >> 5;
    int row  = blockIdx.x * 8 + wave;               // 0 .. R*B*N-1
    const int* p = adj + (size_t)row * N_;
    int s = 0;
#pragma unroll
    for (int t = 0; t < 8; ++t) {
        v4i v = *(const v4i*)(p + lane * 4 + t * 128);
        s += v.x + v.y + v.z + v.w;
    }
#pragma unroll
    for (int off = 16; off > 0; off >>= 1) s += __shfl_down(s, off, 32);
    if (lane == 0) deg[row] = (float)s;
}

// ---------------------------------------------------------------------------
// K2: fused GEMM  [B*N,64] x {wfull_r (r=0..7), Wo}  via v_wmma_f32_16x16x32_f16
//     One wave per 16x16 tile.  A = features rows (K-contiguous f32 -> f16).
//     B = wfullT/WoT rows (K-contiguous f16).
//     r<8 tiles  -> xT[r][b][o][n]  (f16, transposed: A-operand for K3)
//     self tiles -> selfx[b][n][o]  (f32)
// ---------------------------------------------------------------------------
__global__ void k_xself(const float* __restrict__ feat, const _Float16* __restrict__ wfullT,
                        const _Float16* __restrict__ WoT,
                        _Float16* __restrict__ xT, float* __restrict__ selfx) {
    int wid  = blockIdx.x * 8 + (threadIdx.x >> 5);   // 0..9215
    int lane = threadIdx.x & 31;
    int rowTile = wid & 255;                          // 256 row tiles over B*N
    int colTile = wid >> 8;                           // 0..35 (32 = R*O/16, 4 self)
    int row0 = rowTile * 16;
    int l15  = lane & 15;
    int hi   = lane >> 4;
    int kb   = hi * 8;                                // A-operand K base (0 / 8)
    int Kb16 = hi * 16;                               // unused; B-operand also uses kb (0 / 8)

    bool isSelf = (colTile >= 32);
    int otile;
    const _Float16* Bmat;
    if (isSelf) { otile = (colTile - 32) * 16; Bmat = WoT; }
    else        { int r = colTile >> 2; otile = (colTile & 3) * 16;
                  Bmat = wfullT + r * O_ * I_; }

    const float* arow = feat + (size_t)(row0 + l15) * I_;
    v8f c = {};
#pragma unroll
    for (int ks = 0; ks < 2; ++ks) {
        int i0 = ks * 32;
        float fa[16];
        *(v4f*)(fa + 0)  = *(const v4f*)(arow + i0 + kb);
        *(v4f*)(fa + 4)  = *(const v4f*)(arow + i0 + kb + 4);
        *(v4f*)(fa + 8)  = *(const v4f*)(arow + i0 + 16 + kb);
        *(v4f*)(fa + 12) = *(const v4f*)(arow + i0 + 16 + kb + 4);
        v16h a;
#pragma unroll
        for (int e = 0; e < 16; ++e) a[e] = (_Float16)fa[e];

        v8h b0 = *(const v8h*)(Bmat + (otile + l15) * I_ + i0 + kb);
        v8h b1 = *(const v8h*)(Bmat + (otile + l15) * I_ + i0 + 16 + kb);
        v16h bv;
#pragma unroll
        for (int e = 0; e < 8; ++e) { bv[e] = b0[e]; bv[8 + e] = b1[e]; }

        c = __builtin_amdgcn_wmma_f32_16x16x32_f16(false, a, false, bv,
                                                   (short)0, c, false, false);
    }

    int Mb = hi * 8;                                  // D rows g -> M = g + 8*hi
    if (!isSelf) {
        int r = colTile >> 2;
        int b = row0 >> 10, n0 = row0 & 1023;
        v8h h;
#pragma unroll
        for (int g = 0; g < 8; ++g) h[g] = (_Float16)c[g];
        _Float16* dst = xT + ((size_t)(r * B_ + b) * O_ + otile + l15) * N_ + n0 + Mb;
        *(v8h*)dst = h;                               // one 16B store per lane
    } else {
        float* dst = selfx + (size_t)(row0 + Mb) * O_ + otile + l15;
#pragma unroll
        for (int g = 0; g < 8; ++g) dst[g * O_] = c[g];
    }
}

// ---------------------------------------------------------------------------
// K3: main pass.  Per (r,b): D[o,n] = sum_m xT[o,m] * adj[m,n]  (f16 WMMA,
//     f32 acc).  A = xT (contiguous b128 loads).  B = adj column gather:
//     dword loads, lanes 0..15 cover 16 consecutive n -> 64B coalesced.
//     Each wave owns one 16-wide n subtile and ALL 64 o columns (4 WMMAs per
//     k-step sharing one B operand) -> every adj element is loaded exactly
//     once per workgroup; adj L2 traffic = the 128 MB minimum.  No LDS, no
//     barriers.  Fold 1/(deg+eps) into the store.
//     agg layout [R*B][N][O] f32 (o-contiguous -> lane stores 2x float4).
// ---------------------------------------------------------------------------
__global__ void k_agg(const int* __restrict__ adj, const _Float16* __restrict__ xT,
                      const float* __restrict__ deg, float* __restrict__ agg) {
    int blk  = blockIdx.x;            // 0..255
    int rb   = blk >> 3;              // r*B + b
    int n0   = (blk & 7) * 128;       // 128-wide n slab per block
    int wave = threadIdx.x >> 5;      // 0..7 = n subtile within slab
    int lane = threadIdx.x & 31;
    int l15  = lane & 15;
    int hi   = lane >> 4;
    int kb   = hi * 8;                // A K-base (0 / 8)
    int Kb16 = hi * 16;               // unused; B also uses kb (0 / 8)

    int ncol = n0 + wave * 16 + l15;  // this lane's fixed output column n
    const int*      adjc = adj + (size_t)rb * N_ * N_ + ncol;
    const _Float16* xb   = xT + (size_t)rb * O_ * N_ + (size_t)l15 * N_;

    v8f c[4] = {{}, {}, {}, {}};
    for (int m0 = 0; m0 < N_; m0 += 32) {
        // B operand: adj[m0+kb+e+(e&8)][ncol]; adj in {0,1} -> pack 2 f16 lanes
        // with (a0 + (a1<<16)) * 0x3C00  (2 VALU per pair, exact)
        union { v16h h; unsigned u[8]; } bv;
#pragma unroll
        for (int e = 0; e < 16; e += 2) {
            unsigned a0 = (unsigned)adjc[(size_t)(m0 + kb + e + (e & 8))     * N_];
            unsigned a1 = (unsigned)adjc[(size_t)(m0 + kb + e + (e & 8) + 1) * N_];
            bv.u[e >> 1] = (a0 + (a1 << 16)) * 0x3C00u;
        }
        // 4 o-tiles share this B operand
#pragma unroll
        for (int ot = 0; ot < 4; ++ot) {
            const _Float16* xp = xb + (size_t)(ot * 16) * N_ + m0 + kb;
            v8h lo = *(const v8h*)xp;
            v8h hp = *(const v8h*)(xp + 16);
            v16h av;
#pragma unroll
            for (int e = 0; e < 8; ++e) { av[e] = lo[e]; av[8 + e] = hp[e]; }
            c[ot] = __builtin_amdgcn_wmma_f32_16x16x32_f16(false, av, false, bv.h,
                                                           (short)0, c[ot], false, false);
        }
    }

    float s = 1.0f / (deg[(size_t)rb * N_ + ncol] + EPS);
    float* base = agg + ((size_t)rb * N_ + ncol) * O_;   // fixed n per lane
#pragma unroll
    for (int ot = 0; ot < 4; ++ot) {
        v4f w;
#pragma unroll
        for (int g = 0; g < 4; ++g) w[g] = c[ot][g] * s;
        *(v4f*)(base + ot * 16 + hi * 8) = w;
#pragma unroll
        for (int g = 0; g < 4; ++g) w[g] = c[ot][g + 4] * s;
        *(v4f*)(base + ot * 16 + hi * 8 + 4) = w;
    }
}

// ---------------------------------------------------------------------------
// K4: faithful "scrambled" reduce: out[b,n,o] = relu(bias[o] +
//       sum_{j=0..8} concat[b,n,9o+j]),  concat = [agg_r0|...|agg_r7|self]
// ---------------------------------------------------------------------------
__global__ void k_out(const float* __restrict__ agg, const float* __restrict__ selfx,
                      const float* __restrict__ bias, float* __restrict__ out) {
    int tid = blockIdx.x * 256 + threadIdx.x;      // < B*N*O
    int o  = tid & 63;
    int bn = tid >> 6;                             // b*N + n
    int b  = bn >> 10, n = bn & 1023;
    float s = bias[o];
#pragma unroll
    for (int j = 0; j < 9; ++j) {
        int k = 9 * o + j;
        float v;
        if (k < (R_ * O_)) {
            int r = k >> 6, oc = k & 63;
            v = agg[((size_t)(r * B_ + b) * N_ + n) * O_ + oc];
        } else {
            v = selfx[(size_t)bn * O_ + (k - R_ * O_)];
        }
        s += v;
    }
    out[(size_t)bn * O_ + o] = s > 0.f ? s : 0.f;
}

// ---------------------------------------------------------------------------
// Workspace layout (bytes):
//   wfullT f16 [R][O][I]        @ 0        (65536)
//   WoT    f16 [O][I]           @ 65536    (8192)
//   xT     f16 [R][B][O][N]     @ 73728    (4194304)
//   selfx  f32 [B][N][O]        @ 4268032  (1048576)
//   deg    f32 [R][B][N]        @ 5316608  (131072)
//   agg    f32 [R][B][N][O]     @ 5447680  (8388608)   total ~13.8 MB
// ---------------------------------------------------------------------------
extern "C" void kernel_launch(void* const* d_in, const int* in_sizes, int n_in,
                              void* d_out, int out_size, void* d_ws, size_t ws_size,
                              hipStream_t stream) {
    const float* feat   = (const float*)d_in[0];
    const int*   adj    = (const int*)  d_in[1];
    const float* weight = (const float*)d_in[2];
    const float* Wo     = (const float*)d_in[3];
    const float* w_cmp  = (const float*)d_in[4];
    const float* bias   = (const float*)d_in[5];

    char* ws = (char*)d_ws;
    _Float16* wfullT = (_Float16*)(ws + 0);
    _Float16* WoT    = (_Float16*)(ws + 65536);
    _Float16* xT     = (_Float16*)(ws + 73728);
    float*    selfx  = (float*)   (ws + 4268032);
    float*    deg    = (float*)   (ws + 5316608);
    float*    agg    = (float*)   (ws + 5447680);
    float*    out    = (float*)   d_out;

    k_wprep<<<144,  256, 0, stream>>>(weight, Wo, w_cmp, wfullT, WoT);
    k_deg  <<<4096, 256, 0, stream>>>(adj, deg);          // streams adj -> L2
    k_xself<<<1152, 256, 0, stream>>>(feat, wfullT, WoT, xT, selfx);
    k_agg  <<<256,  256, 0, stream>>>(adj, xT, deg, agg); // adj hits L2, read once
    k_out  <<<1024, 256, 0, stream>>>(agg, selfx, bias, out);
}
